// GATCharge_89584427860008
// MI455X (gfx1250) — compile-verified, matrix operand order checked by reading, not run on hardware
//
#include <hip/hip_runtime.h>
#include <hip/hip_bf16.h>

#define GN   100000          // nodes
#define GE   1600000         // edges
#define GC   64              // hidden
#define GET  (GE + GN)       // edges incl. self loops

typedef __attribute__((ext_vector_type(16))) _Float16 v16h;
typedef __attribute__((ext_vector_type(8)))  float    v8f;

// ---------- helpers ----------
__device__ __forceinline__ unsigned enc_ord(float f) {
    unsigned u = __float_as_uint(f);
    return (u & 0x80000000u) ? ~u : (u | 0x80000000u);
}
__device__ __forceinline__ float dec_ord(unsigned u) {
    return (u & 0x80000000u) ? __uint_as_float(u ^ 0x80000000u)
                             : __uint_as_float(~u);
}

// ---------- fills ----------
__global__ void gat_fill_f(float* __restrict__ p, float v, long n) {
    long i = (long)blockIdx.x * blockDim.x + threadIdx.x;
    if (i < n) p[i] = v;
}
__global__ void gat_fill_u(unsigned* __restrict__ p, unsigned v, long n) {
    long i = (long)blockIdx.x * blockDim.x + threadIdx.x;
    if (i < n) p[i] = v;
}

// ---------- self-loop attr (mean of incoming edge_attr) ----------
__global__ void gat_degree(const int* __restrict__ dst, const float* __restrict__ ea,
                           int* __restrict__ cnt, float* __restrict__ la) {
    int e = blockIdx.x * 256 + threadIdx.x;
    if (e >= GE) return;
    int d = dst[e];
    atomicAdd(&cnt[d], 1);
    atomicAdd(&la[2 * d],     ea[2 * e]);
    atomicAdd(&la[2 * d + 1], ea[2 * e + 1]);
}
__global__ void gat_loop_mean(const int* __restrict__ cnt, float* __restrict__ la) {
    int i = blockIdx.x * 256 + threadIdx.x;
    if (i >= GN) return;
    float c = (float)max(cnt[i], 1);
    la[2 * i]     /= c;
    la[2 * i + 1] /= c;
}

// ---------- per-layer folded params: Wa_s, Wa_d (64 each), wea (2) ----------
__global__ void gat_layer_params(const float* __restrict__ W,  const float* __restrict__ a_s,
                                 const float* __restrict__ a_d, const float* __restrict__ We,
                                 const float* __restrict__ a_e, float* __restrict__ out) {
    int t = threadIdx.x;             // 64 threads
    float s1 = 0.f, s2 = 0.f;
    for (int c = 0; c < GC; ++c) {
        float w = W[t * GC + c];
        s1 += w * a_s[c];
        s2 += w * a_d[c];
    }
    out[t] = s1;
    out[64 + t] = s2;
    if (t < 2) {
        float s = 0.f;
        for (int c = 0; c < GC; ++c) s += We[t * GC + c] * a_e[c];
        out[128 + t] = s;
    }
}

// ---------- WMMA GEMM: out[nrows,64] = A[nrows,64] @ W[64,64] (hi/lo f16 split) ----------
__global__ __launch_bounds__(256)
void gat_gemm64_wmma(const float* __restrict__ A, const float* __restrict__ W,
                     float* __restrict__ out, const float* __restrict__ bias,
                     int nrows, int fuse_bias_relu) {
    __shared__ float ldsW[64 * 64];
    for (int i = threadIdx.x; i < 64 * 64; i += 256) ldsW[i] = W[i];
    __syncthreads();

    int wid  = threadIdx.x >> 5;
    int lane = threadIdx.x & 31;
    int tile = blockIdx.x * 8 + wid;
    int ntiles = nrows >> 4;            // nrows divisible by 16
    if (tile >= ntiles) return;

    int m0   = tile << 4;
    int row  = m0 + (lane & 15);
    int ksel = (lane >> 4) << 3;        // lanes 0-15: K base 0, lanes 16-31: K base 8
    const float* ar = A + (long)row * GC;

    // A fragments for both K-steps (hi/lo f16 split of f32)
    v16h ahi[2], alo[2];
    #pragma unroll
    for (int ks = 0; ks < 2; ++ks) {
        #pragma unroll
        for (int h = 0; h < 16; ++h) {
            int k = ks * 32 + ksel + h + (h & 8);
            float v = ar[k];
            _Float16 hi = (_Float16)v;
            ahi[ks][h] = hi;
            alo[ks][h] = (_Float16)(v - (float)hi);
        }
    }

    int nloc = lane & 15;
    #pragma unroll
    for (int nt = 0; nt < 4; ++nt) {
        int n0 = nt << 4;
        v8f acc = {};
        #pragma unroll
        for (int ks = 0; ks < 2; ++ks) {
            v16h bhi, blo;
            #pragma unroll
            for (int h = 0; h < 16; ++h) {
                int k = ks * 32 + ksel + h + (h & 8);
                float v = ldsW[k * 64 + n0 + nloc];
                _Float16 hi = (_Float16)v;
                bhi[h] = hi;
                blo[h] = (_Float16)(v - (float)hi);
            }
            acc = __builtin_amdgcn_wmma_f32_16x16x32_f16(false, ahi[ks], false, bhi, (short)0, acc, false, false);
            acc = __builtin_amdgcn_wmma_f32_16x16x32_f16(false, ahi[ks], false, blo, (short)0, acc, false, false);
            acc = __builtin_amdgcn_wmma_f32_16x16x32_f16(false, alo[ks], false, bhi, (short)0, acc, false, false);
        }
        int col = n0 + nloc;
        float b = fuse_bias_relu ? bias[col] : 0.0f;
        #pragma unroll
        for (int v = 0; v < 8; ++v) {
            int m = m0 + v + ((lane >> 4) << 3);
            float val = acc[v] + b;
            if (fuse_bias_relu) val = fmaxf(val, 0.0f);
            out[(long)m * GC + col] = val;
        }
    }
}

// ---------- per-node dual row-dot: hs[n]=x[n]·v1, hd[n]=x[n]·v2 (wave per node) ----------
__global__ __launch_bounds__(256)
void gat_row_dots(const float* __restrict__ X, const float* __restrict__ v1,
                  const float* __restrict__ v2, float* __restrict__ hs,
                  float* __restrict__ hd) {
    int node = (blockIdx.x * 256 + threadIdx.x) >> 5;
    int lane = threadIdx.x & 31;
    if (node >= GN) return;
    const float* xr = X + (long)node * GC;
    float a0 = xr[2 * lane], a1 = xr[2 * lane + 1];
    float p1 = a0 * v1[2 * lane] + a1 * v1[2 * lane + 1];
    float p2 = a0 * v2[2 * lane] + a1 * v2[2 * lane + 1];
    #pragma unroll
    for (int o = 16; o; o >>= 1) {
        p1 += __shfl_xor(p1, o, 32);
        p2 += __shfl_xor(p2, o, 32);
    }
    if (lane == 0) { hs[node] = p1; hd[node] = p2; }
}

// ---------- attention logits + segment max ----------
__global__ void gat_edge_alpha(const int* __restrict__ src, const int* __restrict__ dst,
                               const float* __restrict__ ea, const float* __restrict__ la,
                               const float* __restrict__ hs, const float* __restrict__ hd,
                               const float* __restrict__ wea, float* __restrict__ alpha,
                               unsigned* __restrict__ amax) {
    int e = blockIdx.x * 256 + threadIdx.x;
    if (e >= GET) return;
    int s, d; float e0, e1;
    if (e < GE) { s = src[e]; d = dst[e]; e0 = ea[2 * e]; e1 = ea[2 * e + 1]; }
    else        { s = d = e - GE; e0 = la[2 * (e - GE)]; e1 = la[2 * (e - GE) + 1]; }
    float a = hs[s] + hd[d] + e0 * wea[0] + e1 * wea[1];
    a = a > 0.f ? a : 0.2f * a;               // leaky_relu(0.2)
    alpha[e] = a;
    atomicMax(&amax[d], enc_ord(a));
}

// ---------- exp + segment denom ----------
__global__ void gat_edge_exp(const int* __restrict__ dst, float* __restrict__ alpha,
                             const unsigned* __restrict__ amax, float* __restrict__ denom) {
    int e = blockIdx.x * 256 + threadIdx.x;
    if (e >= GET) return;
    int d = (e < GE) ? dst[e] : (e - GE);
    float ex = __expf(alpha[e] - dec_ord(amax[d]));
    alpha[e] = ex;
    atomicAdd(&denom[d], ex);
}

// ---------- weighted message scatter-add (wave per edge) ----------
__global__ __launch_bounds__(256)
void gat_edge_msg(const int* __restrict__ src, const int* __restrict__ dst,
                  const float* __restrict__ h, const float* __restrict__ alpha,
                  const float* __restrict__ denom, float* __restrict__ y) {
    int e    = (blockIdx.x * 256 + threadIdx.x) >> 5;
    int lane = threadIdx.x & 31;
    if (e >= GET) return;
    int s = (e < GE) ? src[e] : (e - GE);
    int d = (e < GE) ? dst[e] : (e - GE);
    float w = alpha[e] / denom[d];
    const float* hr = h + (long)s * GC;
    float*       yr = y + (long)d * GC;
    atomicAdd(&yr[lane],      hr[lane]      * w);
    atomicAdd(&yr[lane + 32], hr[lane + 32] * w);
}

// ---------- elementwise ----------
__global__ void gat_bias_relu(float* __restrict__ y, const float* __restrict__ b) {
    long i = (long)blockIdx.x * 256 + threadIdx.x;
    if (i >= (long)GN * GC) return;
    int c = (int)(i & 63);
    y[i] = fmaxf(y[i] + b[c], 0.0f);
}
__global__ void gat_vadd(const float* __restrict__ a, const float* __restrict__ b,
                         float* __restrict__ o, long n) {
    long i = (long)blockIdx.x * 256 + threadIdx.x;
    if (i < n) o[i] = a[i] + b[i];
}

// ---------- final head: xr[n] = y4[n]·lf_W + lf_b (wave per node) ----------
__global__ __launch_bounds__(256)
void gat_final_dot(const float* __restrict__ y4, const float* __restrict__ w,
                   const float* __restrict__ b, float* __restrict__ xr) {
    int node = (blockIdx.x * 256 + threadIdx.x) >> 5;
    int lane = threadIdx.x & 31;
    if (node >= GN) return;
    const float* r = y4 + (long)node * GC;
    float p = r[2 * lane] * w[2 * lane] + r[2 * lane + 1] * w[2 * lane + 1];
    #pragma unroll
    for (int o = 16; o; o >>= 1) p += __shfl_xor(p, o, 32);
    if (lane == 0) xr[node] = p + b[0];
}

extern "C" void kernel_launch(void* const* d_in, const int* in_sizes, int n_in,
                              void* d_out, int out_size, void* d_ws, size_t ws_size,
                              hipStream_t stream) {
    const float* x   = (const float*)d_in[0];
    const int*   ei  = (const int*)d_in[1];
    const float* ea  = (const float*)d_in[2];
    // d_in[3] = shift (unused by reference)
    const float* gW [3] = {(const float*)d_in[4],  (const float*)d_in[10], (const float*)d_in[16]};
    const float* gAs[3] = {(const float*)d_in[5],  (const float*)d_in[11], (const float*)d_in[17]};
    const float* gAd[3] = {(const float*)d_in[6],  (const float*)d_in[12], (const float*)d_in[18]};
    const float* gAe[3] = {(const float*)d_in[7],  (const float*)d_in[13], (const float*)d_in[19]};
    const float* gWe[3] = {(const float*)d_in[8],  (const float*)d_in[14], (const float*)d_in[20]};
    const float* gB [3] = {(const float*)d_in[9],  (const float*)d_in[15], (const float*)d_in[21]};
    const float* l2W = (const float*)d_in[22];
    const float* l2b = (const float*)d_in[23];
    const float* lfW = (const float*)d_in[24];
    const float* lfb = (const float*)d_in[25];

    const int* src = ei;          // edge_index row 0
    const int* dst = ei + GE;     // edge_index row 1

    // workspace carve-up (floats)
    const long NC = (long)GN * GC;
    float* ws     = (float*)d_ws;
    float* h      = ws;                 // [N,64]  (also reused as y4)
    float* y0     = h   + NC;           // [N,64]
    float* y1     = y0  + NC;           // [N,64]
    float* t      = y1  + NC;           // [N,64]  (y0+y1, then y0+y1+y3)
    float* alpha  = t   + NC;           // [Et]
    float* hs     = alpha + GET;        // [N]
    float* hd     = hs  + GN;           // [N]
    unsigned* amax = (unsigned*)(hd + GN); // [N]
    float* denom  = hd  + 2 * (long)GN; // [N]  (after amax)
    int*   cnt    = (int*)(denom + GN); // [N]
    float* la     = (float*)(cnt + GN); // [N,2]
    float* params = la + 2 * (long)GN;  // 3 * 144

    float* xr = (float*)d_out;          // [N]
    float* y3 = (float*)d_out + GN;     // [N,64]

    const int TB = 256;
    dim3 b(TB);
    int gE   = (GET + TB - 1) / TB;                 // thread-per-edge grids
    int gEw  = (int)(((long)GET * 32 + TB - 1) / TB); // wave-per-edge
    int gN   = (GN + TB - 1) / TB;
    int gNw  = (int)(((long)GN * 32 + TB - 1) / TB);  // wave-per-node
    int gNC  = (int)((NC + TB - 1) / TB);
    int gGemm = ((GN >> 4) + 7) / 8;                // 8 M-tiles of 16 per block

    // ---- self-loop attrs ----
    gat_fill_u<<<gN, b, 0, stream>>>((unsigned*)cnt, 0u, GN);
    gat_fill_f<<<(2 * GN + TB - 1) / TB, b, 0, stream>>>(la, 0.f, 2L * GN);
    gat_degree<<<(GE + TB - 1) / TB, b, 0, stream>>>(dst, ea, cnt, la);
    gat_loop_mean<<<gN, b, 0, stream>>>(cnt, la);

    // ---- folded attention params per layer ----
    for (int l = 0; l < 3; ++l)
        gat_layer_params<<<1, 64, 0, stream>>>(gW[l], gAs[l], gAd[l], gWe[l], gAe[l],
                                               params + l * 144);

    const float* xin[3] = {x, y0, t};
    float* yout[3]      = {y0, y1, y3};

    for (int l = 0; l < 3; ++l) {
        if (l == 2)  // layer-3 input = y0 + y1
            gat_vadd<<<gNC, b, 0, stream>>>(y0, y1, t, NC);

        // dense part: h = xin @ W  (WMMA)
        gat_gemm64_wmma<<<gGemm, b, 0, stream>>>(xin[l], gW[l], h, nullptr, GN, 0);
        // per-node attention scalars: hs = xin·(W a_s), hd = xin·(W a_d)
        gat_row_dots<<<gNw, b, 0, stream>>>(xin[l], params + l * 144,
                                            params + l * 144 + 64, hs, hd);

        gat_fill_u<<<gN, b, 0, stream>>>(amax, 0u, GN);
        gat_fill_f<<<gN, b, 0, stream>>>(denom, 0.f, GN);
        gat_fill_f<<<gNC, b, 0, stream>>>(yout[l], 0.f, NC);

        gat_edge_alpha<<<gE, b, 0, stream>>>(src, dst, ea, la, hs, hd,
                                             params + l * 144 + 128, alpha, amax);
        gat_edge_exp<<<gE, b, 0, stream>>>(dst, alpha, amax, denom);
        gat_edge_msg<<<gEw, b, 0, stream>>>(src, dst, h, alpha, denom, yout[l]);
        gat_bias_relu<<<gNC, b, 0, stream>>>(yout[l], gB[l]);
    }

    // s = (y0+y1) + y3 ; y4 = relu(s @ l2_W + l2_b) ; xr = y4 @ lf_W + lf_b
    gat_vadd<<<gNC, b, 0, stream>>>(t, y3, t, NC);
    gat_gemm64_wmma<<<gGemm, b, 0, stream>>>(t, l2W, h, l2b, GN, 1);
    gat_final_dot<<<gNw, b, 0, stream>>>(h, lfW, lfb, xr);
}
